// Head_60516089201055
// MI455X (gfx1250) — matrix operand order repaired and checked by reading, not compiled, not run on hardware
//
#include <hip/hip_runtime.h>

// ---------------------------------------------------------------------------
// Causal single-head attention for MI455X (gfx1250, wave32, WMMA).
//   B=256, T=256, C=1024, H=64
// Phase 1 (fused): k,q,v = x @ {Wk,Wq,Wv} in ONE pass over x (x is the
//   dominant HBM traffic: 268 MB; reading it once puts us at the ~12us
//   roofline floor @ 23.3 TB/s). All three W chunks staged in LDS as f16,
//   each x A-fragment feeds 12 v_wmma_f32_16x16x32_f16.
// Phase 2: flash-attention (online softmax) with WMMA; q/k/vT live in a
//   24 MB f16 workspace -> fully resident in the 192 MB L2.
// ---------------------------------------------------------------------------

typedef __attribute__((ext_vector_type(16))) _Float16 v16h;
typedef __attribute__((ext_vector_type(8)))  _Float16 v8h;
typedef __attribute__((ext_vector_type(8)))  float    v8f;

#define B_ 256
#define T_ 256
#define C_ 1024
#define H_ 64
#define CCHUNK 128  // C columns staged per LDS round (3*64*128*2B = 48 KB)

static __device__ __forceinline__ v8f wmma_f16f32(v16h a, v16h b, v8f c) {
  // 8-arg form: (neg_a, A, neg_b, B, c_mod, C, reuse_a, reuse_b)
  return __builtin_amdgcn_wmma_f32_16x16x32_f16(false, a, false, b, (short)0, c,
                                                false, false);
}

// ---------------------------------------------------------------------------
// Phase 1: fused projections. grid = 512, block = 256 (8 waves).
// Each wave owns one 16-row tile of the flattened (B*T) x C input and
// produces 16x64 f16 tiles for k, q and v simultaneously (12 accumulators).
// W chunks (3 x 64 x 128, f16, transposed) staged in LDS so B-fragments are
// contiguous ds_load_b128 reads. W re-reads per block hit L2 (768 KB total).
// ---------------------------------------------------------------------------
__global__ __launch_bounds__(256) void proj_kernel(
    const float* __restrict__ x,
    const float* __restrict__ Wk, const float* __restrict__ Wq,
    const float* __restrict__ Wv,
    _Float16* __restrict__ kf, _Float16* __restrict__ qf,
    _Float16* __restrict__ vT) {
  __shared__ __align__(32) _Float16 wlds[3 * H_ * CCHUNK];  // 48 KB

  const int tid  = threadIdx.x;
  const int wave = tid >> 5;
  const int lane = tid & 31;
  const int n    = lane & 15;   // M (A) or N (B/C/D) index
  const int hf   = lane >> 4;   // lane half

  const float* Wm[3] = {Wk, Wq, Wv};

  const int tile = blockIdx.x * 8 + wave;   // 16-row tile over B*T (4096 tiles)
  const int rt0  = tile * 16;

  v8f acc[3][4] = {};  // {k,q,v} x (16 rows x 64 cols as 4 fragments), f32

  for (int c0 = 0; c0 < C_; c0 += CCHUNK) {
    __syncthreads();
    // Stage all three W[c0:c0+128, 0:64] transposed into LDS as f16.
    for (int idx = tid; idx < 3 * H_ * CCHUNK; idx += 256) {
      const int mtx = idx / (H_ * CCHUNK);
      const int rem = idx - mtx * (H_ * CCHUNK);
      const int h   = rem & 63;
      const int c   = rem >> 6;
      wlds[mtx * (H_ * CCHUNK) + h * CCHUNK + c] =
          (_Float16)Wm[mtx][(size_t)(c0 + c) * H_ + h];
    }
    __syncthreads();

    const float* xrow = x + (size_t)(rt0 + n) * C_ + c0;

    for (int kc = 0; kc < CCHUNK; kc += 32) {
      // A-fragment: 16x32 f16 of x (ISA layout: lane half selects K groups
      // {8h..8h+7} and {16+8h..16+8h+7}); two contiguous float4 pairs each.
      const float4* r0 = reinterpret_cast<const float4*>(xrow + kc + 8 * hf);
      const float4* r1 = reinterpret_cast<const float4*>(xrow + kc + 16 + 8 * hf);
      const float4 f0 = r0[0], f1 = r0[1], f2 = r1[0], f3 = r1[1];
      v16h a;
      a[0]  = (_Float16)f0.x; a[1]  = (_Float16)f0.y;
      a[2]  = (_Float16)f0.z; a[3]  = (_Float16)f0.w;
      a[4]  = (_Float16)f1.x; a[5]  = (_Float16)f1.y;
      a[6]  = (_Float16)f1.z; a[7]  = (_Float16)f1.w;
      a[8]  = (_Float16)f2.x; a[9]  = (_Float16)f2.y;
      a[10] = (_Float16)f2.z; a[11] = (_Float16)f2.w;
      a[12] = (_Float16)f3.x; a[13] = (_Float16)f3.y;
      a[14] = (_Float16)f3.z; a[15] = (_Float16)f3.w;

      // B-fragments from LDS: lane = N, K groups {8h..8h+7},{16+8h..16+8h+7}.
      // One converted x fragment feeds 12 WMMAs (k, q and v tiles).
      const int bbase = kc + 8 * hf;
#pragma unroll
      for (int m = 0; m < 3; ++m)
#pragma unroll
        for (int f = 0; f < 4; ++f) {
          const _Float16* wp = &wlds[m * (H_ * CCHUNK) + (f * 16 + n) * CCHUNK + bbase];
          const v8h w0 = *reinterpret_cast<const v8h*>(wp);
          const v8h w1 = *reinterpret_cast<const v8h*>(wp + 16);
          v16h bf;
#pragma unroll
          for (int i = 0; i < 8; ++i) { bf[i] = w0[i]; bf[8 + i] = w1[i]; }
          acc[m][f] = wmma_f16f32(a, bf, acc[m][f]);
        }
    }
  }

  // Store f16 results. C/D layout: row = r + 8*hf, col = 16*f + n.
  {
    _Float16* dst[2] = {kf, qf};
#pragma unroll
    for (int m = 0; m < 2; ++m)
#pragma unroll
      for (int f = 0; f < 4; ++f)
#pragma unroll
        for (int r = 0; r < 8; ++r)
          dst[m][(size_t)(rt0 + r + 8 * hf) * H_ + f * 16 + n] =
              (_Float16)acc[m][f][r];
  }
  {
    const int b  = rt0 / T_;
    const int t0 = rt0 % T_;
#pragma unroll
    for (int f = 0; f < 4; ++f)
#pragma unroll
      for (int r = 0; r < 8; ++r)
        vT[((size_t)b * H_ + f * 16 + n) * T_ + t0 + r + 8 * hf] =
            (_Float16)acc[2][f][r];
  }
}

// ---------------------------------------------------------------------------
// Phase 2: causal flash-attention. grid = 512, block = 256 (8 waves).
// One wave per (batch, 16-row tile): online softmax over 32-wide column
// super-blocks; S = q k^T via WMMA, P converted D-layout -> A-layout through
// per-wave LDS, O += P V via WMMA.
// ---------------------------------------------------------------------------
__global__ __launch_bounds__(256) void attn_kernel(
    const _Float16* __restrict__ qf, const _Float16* __restrict__ kf,
    const _Float16* __restrict__ vT, float* __restrict__ out) {
  __shared__ __align__(32) _Float16 plds[8 * 16 * 32];  // per-wave P staging

  const int tid  = threadIdx.x;
  const int wave = tid >> 5;
  const int lane = tid & 31;
  const int n    = lane & 15;
  const int hf   = lane >> 4;
  _Float16* pw   = &plds[wave * 512];

  const int tile = blockIdx.x * 8 + wave;  // over B * (T/16) = 4096
  const int b    = tile >> 4;
  const int t0   = (tile & 15) * 16;

  // q A-fragments (2 chunks of K=32 covering H=64).
  v16h qa[2];
  const _Float16* qrow = qf + ((size_t)b * T_ + t0 + n) * H_;
#pragma unroll
  for (int c = 0; c < 2; ++c) {
    const v8h u0 = *reinterpret_cast<const v8h*>(qrow + 32 * c + 16 * hf);
    const v8h u1 = *reinterpret_cast<const v8h*>(qrow + 32 * c + 16 * hf + 8);
#pragma unroll
    for (int i = 0; i < 8; ++i) {
      qa[c][i]     = u0[i];
      qa[c][8 + i] = u1[i];
    }
  }

  float m[8], lsum[8];
#pragma unroll
  for (int r = 0; r < 8; ++r) {
    m[r]    = -1e30f;
    lsum[r] = 0.0f;
  }
  v8f oacc[4] = {};  // 16 x 64 f32 output accumulator

  const float scale = 0.125f;  // 1/sqrt(H), H = 64

  for (int s0 = 0; s0 <= t0 + 15; s0 += 32) {
    if (s0 + 32 <= t0 + 15)  // hint next k tile toward the caches
      __builtin_prefetch(kf + ((size_t)b * T_ + s0 + 32 + n) * H_, 0, 3);

    float sv[2][8];
#pragma unroll
    for (int sb = 0; sb < 2; ++sb) {
      const int s = s0 + 16 * sb;
      if (s <= t0 + 15) {  // wave-uniform branch (EXEC stays all-1s for WMMA)
        v8f sacc = {};
#pragma unroll
        for (int c = 0; c < 2; ++c) {
          const v16h kb = *reinterpret_cast<const v16h*>(
              kf + ((size_t)b * T_ + s + n) * H_ + 32 * c + 16 * hf);
          sacc = wmma_f16f32(qa[c], kb, sacc);
        }
#pragma unroll
        for (int r = 0; r < 8; ++r) {
          const int row = t0 + r + 8 * hf;
          const int col = s + n;
          sv[sb][r] = (col <= row) ? sacc[r] * scale : -1e30f;
        }
      } else {
#pragma unroll
        for (int r = 0; r < 8; ++r) sv[sb][r] = -1e30f;
      }
    }

    // Online softmax update. Row r+8*hf lives on lanes {16*hf..16*hf+15},
    // so xor-shuffles with masks 1,2,4,8 reduce within each half.
    float alpha[8], p0[8], p1[8];
#pragma unroll
    for (int r = 0; r < 8; ++r) {
      float v = fmaxf(sv[0][r], sv[1][r]);
#pragma unroll
      for (int msk = 1; msk < 16; msk <<= 1) v = fmaxf(v, __shfl_xor(v, msk, 32));
      const float mn = fmaxf(m[r], v);
      alpha[r]       = __expf(m[r] - mn);
      m[r]           = mn;
      p0[r]          = __expf(sv[0][r] - mn);
      p1[r]          = __expf(sv[1][r] - mn);
      float ps = p0[r] + p1[r];
#pragma unroll
      for (int msk = 1; msk < 16; msk <<= 1) ps += __shfl_xor(ps, msk, 32);
      lsum[r] = lsum[r] * alpha[r] + ps;
    }

#pragma unroll
    for (int f = 0; f < 4; ++f)
#pragma unroll
      for (int r = 0; r < 8; ++r) oacc[f][r] *= alpha[r];

    // D-layout P -> LDS -> A-layout P (16x32 f16) for the P@V WMMA.
#pragma unroll
    for (int r = 0; r < 8; ++r) {
      pw[(r + 8 * hf) * 32 + n]      = (_Float16)p0[r];
      pw[(r + 8 * hf) * 32 + 16 + n] = (_Float16)p1[r];
    }
    asm volatile("s_wait_dscnt 0" ::: "memory");  // same-wave RAW on LDS
    v16h pa;
    {
      const v8h u0 = *reinterpret_cast<const v8h*>(&pw[n * 32 + 16 * hf]);
      const v8h u1 = *reinterpret_cast<const v8h*>(&pw[n * 32 + 16 * hf + 8]);
#pragma unroll
      for (int i = 0; i < 8; ++i) {
        pa[i]     = u0[i];
        pa[8 + i] = u1[i];
      }
    }

    // O += P @ V : V stored [B,H,T] so B-fragments are contiguous 32B loads.
#pragma unroll
    for (int f = 0; f < 4; ++f) {
      const v16h vb = *reinterpret_cast<const v16h*>(
          vT + ((size_t)b * H_ + f * 16 + n) * T_ + s0 + 16 * hf);
      oacc[f] = wmma_f16f32(pa, vb, oacc[f]);
    }
  }

  // Epilogue: normalize and store f32 output [B,T,H].
#pragma unroll
  for (int r = 0; r < 8; ++r) {
    const float inv = 1.0f / lsum[r];
#pragma unroll
    for (int f = 0; f < 4; ++f)
      out[((size_t)b * T_ + t0 + r + 8 * hf) * H_ + f * 16 + n] =
          oacc[f][r] * inv;
  }
}

// ---------------------------------------------------------------------------
extern "C" void kernel_launch(void* const* d_in, const int* in_sizes, int n_in,
                              void* d_out, int out_size, void* d_ws,
                              size_t ws_size, hipStream_t stream) {
  (void)in_sizes; (void)n_in; (void)out_size; (void)ws_size;
  const float* x  = (const float*)d_in[0];
  const float* Wk = (const float*)d_in[1];
  const float* Wq = (const float*)d_in[2];
  const float* Wv = (const float*)d_in[3];

  // Workspace: k, q (f16 [B,T,H]) and v transposed (f16 [B,H,T]) = 24 MB.
  _Float16* ws = (_Float16*)d_ws;
  const size_t elems = (size_t)B_ * T_ * H_;
  _Float16* kf = ws;
  _Float16* qf = ws + elems;
  _Float16* vT = ws + 2 * elems;

  const dim3 blk(256);
  const dim3 g1((B_ * T_ / 16) / 8);      // 512, fused k/q/v pass
  const dim3 g2((B_ * (T_ / 16)) / 8);    // 512

  proj_kernel<<<g1, blk, 0, stream>>>(x, Wk, Wq, Wv, kf, qf, vT);
  attn_kernel<<<g2, blk, 0, stream>>>(qf, kf, vT, (float*)d_out);
}
